// FoldingBlock_45801531245211
// MI455X (gfx1250) — matrix operand order repaired and checked by reading, not compiled, not run on hardware
//
#include <hip/hip_runtime.h>

typedef __bf16 bf16_t;
typedef __attribute__((ext_vector_type(16))) __bf16 v16bf;
typedef __attribute__((ext_vector_type(8)))  __bf16 v8bf;
typedef __attribute__((ext_vector_type(8)))  float  v8f;

#define NB   32
#define NQ   2048
#define NIN  512
#define CH   128

// ---- fragment loaders (wave32, 16x16x32 bf16 WMMA) -------------------------
// A fragment (16x32, row-major source): lane%16 = row M; lanes<16 hold K 0..7 &
// 16..23, lanes>=16 hold K 8..15 & 24..31.  p points at row + k0 + (lane/16)*8.
__device__ __forceinline__ v16bf frag_a16x32(const bf16_t* p){
  v8bf lo = *(const v8bf*)p;
  v8bf hi = *(const v8bf*)(p + 16);
  v16bf r;
#pragma unroll
  for (int i = 0; i < 8; ++i){ r[i] = lo[i]; r[i+8] = hi[i]; }
  return r;
}
// B fragment (32x16): lane%16 = col N; lanes<16 hold K 0..7 & 16..23, lanes>=16 K 8..15 & 24..31.
// p points at col-row + k0 + (lane/16)*8 in a K-contiguous layout.
__device__ __forceinline__ v16bf frag_bk16(const bf16_t* p){
  v8bf lo = *(const v8bf*)p;
  v8bf hi = *(const v8bf*)(p + 16);
  v16bf r;
#pragma unroll
  for (int i = 0; i < 8; ++i){ r[i] = lo[i]; r[i+8] = hi[i]; }
  return r;
}

// ---- generic bf16 WMMA GEMM: out = act(A[M,K] @ Wt[N,K]^T + bias) ----------
// 256 threads = 8 waves; block tile 256(M) x 64(N); wave tile 32x64 with 8
// accumulators.  The B tile (64 cols x 32 K, 4 KB) is double-buffered in LDS
// via async global->LDS loads (ASYNCcnt), overlapping fill with WMMA.
// LDS column stride = 80 B (20-bank step) -> conflict-free b128 fragment reads.
__global__ __launch_bounds__(256) void wmma_gemm(
    const bf16_t* __restrict__ A, long lda,
    const bf16_t* __restrict__ Wt, int K,          // Wt is [N,K] (col-major W)
    const float* __restrict__ biasVec,             // [N] or null
    const float* __restrict__ biasMat, int biasMatLd, // [2048,N] by row%2048, or null
    bf16_t* __restrict__ outB, long ldoB,          // bf16 row-major out, or null
    bf16_t* __restrict__ outT, int NcT,            // per-batch transposed out, or null
    float*  __restrict__ outF, long ldoF, int accumF, // f32 out (store/accum), or null
    int relu)
{
  __shared__ bf16_t lds_b[2][64 * 40];             // 2 x 64 cols x 32 K (+8 pad)
  const int wave = threadIdx.x >> 5;
  const int lane = threadIdx.x & 31;
  const int hf   = lane >> 4;
  const int mr   = lane & 15;
  const long m0  = (long)blockIdx.x * 256 + wave * 32;
  const int  n0  = blockIdx.y * 64;

  // cooperative B-tile loader: thread -> (col 0..63, 16B chunk 0..3)
  const int bcol   = threadIdx.x >> 2;
  const int bchunk = threadIdx.x & 3;
  const bf16_t* bsrc = Wt + (long)(n0 + bcol) * K + bchunk * 8;
  const unsigned ldso0 = (unsigned)(unsigned long long)(size_t)&lds_b[0][bcol*40 + bchunk*8];
  const unsigned ldso1 = (unsigned)(unsigned long long)(size_t)&lds_b[1][bcol*40 + bchunk*8];

  v8f acc[2][4] = {};
  const bf16_t* pa0 = A + (m0      + mr) * lda + hf * 8;
  const bf16_t* pa1 = A + (m0 + 16 + mr) * lda + hf * 8;

  const int nk = K >> 5;
  // prime buffer 0
  asm volatile("global_load_async_to_lds_b128 %0, %1, off"
               :: "v"(ldso0),
                  "v"((unsigned long long)(size_t)bsrc)
               : "memory");
  int buf = 0;
  for (int it = 0; it < nk; ++it){
    const int k0 = it << 5;
    const bool pre = (it + 1 < nk);
    if (pre){
      const unsigned dsto = buf ? ldso0 : ldso1;
      asm volatile("global_load_async_to_lds_b128 %0, %1, off"
                   :: "v"(dsto),
                      "v"((unsigned long long)(size_t)(bsrc + k0 + 32))
                   : "memory");
      asm volatile("s_wait_asynccnt 1" ::: "memory");
    } else {
      asm volatile("s_wait_asynccnt 0" ::: "memory");
    }
    __syncthreads();                                   // tile visible to all waves

    v16bf a0 = frag_a16x32(pa0 + k0);
    v16bf a1 = frag_a16x32(pa1 + k0);
    const bf16_t* lb = &lds_b[buf][0];
#pragma unroll
    for (int t = 0; t < 4; ++t){
      v16bf bfrag = frag_bk16(lb + (t*16 + mr)*40 + hf*8);
      acc[0][t] = __builtin_amdgcn_wmma_f32_16x16x32_bf16(
                    false, a0, false, bfrag, (short)0, acc[0][t], false, false);
      acc[1][t] = __builtin_amdgcn_wmma_f32_16x16x32_bf16(
                    false, a1, false, bfrag, (short)0, acc[1][t], false, false);
    }
    __syncthreads();                                   // done reading before reuse
    buf ^= 1;
  }

  // C layout: VGPR i -> row base + i + 8*(lane/16); col n0 + t*16 + lane%16.
#pragma unroll
  for (int rr = 0; rr < 2; ++rr){
#pragma unroll
    for (int t = 0; t < 4; ++t){
      const int colg = n0 + t*16 + mr;
#pragma unroll
      for (int i = 0; i < 8; ++i){
        const long row = m0 + rr*16 + i + 8*hf;
        float v = acc[rr][t][i];
        if (biasVec) v += biasVec[colg];
        if (biasMat) v += biasMat[(row & (NQ-1)) * (long)biasMatLd + colg];
        if (relu)    v = fmaxf(v, 0.0f);
        if (outB) outB[row * ldoB + colg] = (bf16_t)v;
        if (outT){ const long b = row >> 11, n = row & (NQ-1);
                   outT[(b * (long)NcT + colg) * (long)NQ + n] = (bf16_t)v; }
        if (outF){ const long idx = row * ldoF + colg;
                   if (accumF) outF[idx] += v; else outF[idx] = v; }
      }
    }
  }
}

// ---- flash-attention: Ao = qc + softmax(l h^T) @ g  (per batch) ------------
// grid: NB * (NQ/128) blocks, 8 waves/block, each wave owns 16 query rows and a
// 16x256 fp32 O accumulator; streams 32 keys per step with online softmax.
__global__ __launch_bounds__(256) void attn_kernel(
    const bf16_t* __restrict__ Lm,   // [B*NQ, 64]
    const bf16_t* __restrict__ Hm,   // [B*NQ, 64]
    const bf16_t* __restrict__ Gt,   // [B, 256, NQ]  (g transposed)
    const bf16_t* __restrict__ Qc,   // [B*NQ, 256]
    bf16_t* __restrict__ Ao)         // [B*NQ, 256]
{
  __shared__ bf16_t ptile[8][16*32];
  const int wave = threadIdx.x >> 5;
  const int lane = threadIdx.x & 31;
  const int hf = lane >> 4;
  const int mr = lane & 15;
  const int b  = blockIdx.x >> 4;
  const int m0 = (blockIdx.x & 15) * 128 + wave * 16;
  const long rowbase = (long)b * NQ + m0;

  v16bf la[2];
#pragma unroll
  for (int kk = 0; kk < 2; ++kk)
    la[kk] = frag_a16x32(Lm + (rowbase + mr) * 64 + kk*32 + hf*8);

  v8f O[16] = {};
  float rm[8], rs[8];
#pragma unroll
  for (int i = 0; i < 8; ++i){ rm[i] = -3.0e38f; rs[i] = 0.f; }

  for (int j0 = 0; j0 < NQ; j0 += 32){
    v8f s0 = {}, s1 = {};
#pragma unroll
    for (int kk = 0; kk < 2; ++kk){   // logits: h row-major == h^T col-contig
      v16bf hb0 = frag_bk16(Hm + ((long)b*NQ + j0      + mr) * 64 + kk*32 + hf*8);
      v16bf hb1 = frag_bk16(Hm + ((long)b*NQ + j0 + 16 + mr) * 64 + kk*32 + hf*8);
      s0 = __builtin_amdgcn_wmma_f32_16x16x32_bf16(false, la[kk], false, hb0, (short)0, s0, false, false);
      s1 = __builtin_amdgcn_wmma_f32_16x16x32_bf16(false, la[kk], false, hb1, (short)0, s1, false, false);
    }
#pragma unroll
    for (int i = 0; i < 8; ++i){      // online softmax per row (16-lane groups)
      float tmax = fmaxf(s0[i], s1[i]);
      tmax = fmaxf(tmax, __shfl_xor(tmax, 1));
      tmax = fmaxf(tmax, __shfl_xor(tmax, 2));
      tmax = fmaxf(tmax, __shfl_xor(tmax, 4));
      tmax = fmaxf(tmax, __shfl_xor(tmax, 8));
      const float nm = fmaxf(rm[i], tmax);
      const float p0 = __expf(s0[i] - nm);
      const float p1 = __expf(s1[i] - nm);
      float rsum = p0 + p1;
      rsum += __shfl_xor(rsum, 1);
      rsum += __shfl_xor(rsum, 2);
      rsum += __shfl_xor(rsum, 4);
      rsum += __shfl_xor(rsum, 8);
      const float sc = __expf(rm[i] - nm);
      rs[i] = rs[i] * sc + rsum;
      rm[i] = nm;
#pragma unroll
      for (int t = 0; t < 16; ++t) O[t][i] *= sc;
      ptile[wave][(i + 8*hf)*32 + mr]      = (bf16_t)p0;  // C-layout -> LDS
      ptile[wave][(i + 8*hf)*32 + mr + 16] = (bf16_t)p1;
    }
    __syncthreads();
    v16bf pfrag = frag_a16x32(&ptile[wave][mr*32 + hf*8]); // LDS -> A-layout
    __syncthreads();
#pragma unroll
    for (int t = 0; t < 16; ++t){     // O += P @ g  over 256 output columns
      v16bf gb = frag_bk16(Gt + ((long)b*256 + t*16 + mr) * (long)NQ + j0 + hf*8);
      O[t] = __builtin_amdgcn_wmma_f32_16x16x32_bf16(false, pfrag, false, gb, (short)0, O[t], false, false);
    }
  }
#pragma unroll
  for (int i = 0; i < 8; ++i){
    const float inv = 1.0f / rs[i];
    const long row = rowbase + i + 8*hf;
#pragma unroll
    for (int t = 0; t < 16; ++t){
      const int col = t*16 + mr;
      const float v = O[t][i] * inv + (float)Qc[row*256 + col];
      Ao[row*256 + col] = (bf16_t)v;
    }
  }
}

// ---- small helper kernels ---------------------------------------------------
__global__ void k_f32_to_bf16(const float* __restrict__ in, bf16_t* __restrict__ o, long n){
  long i = (long)blockIdx.x*256 + threadIdx.x; if (i < n) o[i] = (bf16_t)in[i];
}
__global__ void k_sub_bf16(const float* __restrict__ a, const float* __restrict__ b,
                           bf16_t* __restrict__ o, long n){
  long i = (long)blockIdx.x*256 + threadIdx.x; if (i < n) o[i] = (bf16_t)(a[i] - b[i]);
}
// out[N,K] = bf16(in[K,N])   (weight transpose to K-contiguous)
__global__ void k_transpose(const float* __restrict__ in, bf16_t* __restrict__ o, int K, int N){
  long i = (long)blockIdx.x*256 + threadIdx.x;
  if (i >= (long)K*N) return;
  long k = i / N, n = i % N;
  o[n*(long)K + k] = (bf16_t)in[i];
}
// grid_term[n,c] = pts[n,0]*Wup[128,c] + pts[n,1]*Wup[129,c] + bup[c]
__global__ void k_gridterm(const float* __restrict__ Wup, const float* __restrict__ bup,
                           float* __restrict__ gt){
  int idx = blockIdx.x*256 + threadIdx.x;
  if (idx >= NQ*CH) return;
  int q = idx >> 7, c = idx & 127;
  int i = q >> 6, j = q & 63;                 // iy over 32, ix over 64
  float iy = rintf(i * (45.0f/31.0f));
  float ix = rintf(j * (45.0f/63.0f));
  float x0 = -0.3f + iy * (0.6f/45.0f);
  float x1 = -0.3f + ix * (0.6f/45.0f);
  gt[idx] = x0*Wup[128*CH + c] + x1*Wup[129*CH + c] + bup[c];
}
// qc[:, :, 0:128] = tile(p, 4) as bf16
__global__ void k_qc_copy(const bf16_t* __restrict__ pb, bf16_t* __restrict__ qc){
  long i = (long)blockIdx.x*256 + threadIdx.x;
  if (i >= (long)NB*NQ*CH) return;
  long m = i >> 7; int c = i & 127;
  long src = (m >> 11)*NIN + (m & (NIN-1));   // (m%2048)%512 == m%512
  qc[m*256 + c] = pb[src*CH + c];
}

// ---- host driver ------------------------------------------------------------
static inline unsigned cdivu(long a, long b){ return (unsigned)((a + b - 1) / b); }

extern "C" void kernel_launch(void* const* d_in, const int* in_sizes, int n_in,
                              void* d_out, int out_size, void* d_ws, size_t ws_size,
                              hipStream_t stream) {
  (void)in_sizes; (void)n_in; (void)out_size; (void)ws_size;
  const float* p   = (const float*)d_in[0];
  const float* Wh  = (const float*)d_in[1];
  const float* bh  = (const float*)d_in[2];
  const float* Wl  = (const float*)d_in[3];
  const float* bl  = (const float*)d_in[4];
  const float* Wg  = (const float*)d_in[5];
  const float* bg  = (const float*)d_in[6];
  const float* Wf  = (const float*)d_in[7];
  const float* bfv = (const float*)d_in[8];
  const float* Wup = (const float*)d_in[9];
  const float* bup = (const float*)d_in[10];
  const float* Wd1 = (const float*)d_in[11];
  const float* bd1 = (const float*)d_in[12];
  const float* Wd2 = (const float*)d_in[13];
  const float* bd2 = (const float*)d_in[14];
  float* out = (float*)d_out;

  size_t off = 0;
  auto alloc = [&](size_t elems, size_t esz)->void*{
    void* r = (char*)d_ws + off; off += (elems*esz + 255) & ~(size_t)255; return r; };
  bf16_t* pb   = (bf16_t*)alloc(2097152, 2);   // bf16(p)               [B*512,128]
  bf16_t* pb2  = (bf16_t*)alloc(2097152, 2);   // bf16(p - down(p1))
  bf16_t* WupT = (bf16_t*)alloc(16384, 2);
  bf16_t* WhT  = (bf16_t*)alloc(16384, 2);
  bf16_t* WlT  = (bf16_t*)alloc(16384, 2);
  bf16_t* WgT  = (bf16_t*)alloc(65536, 2);
  bf16_t* WfT  = (bf16_t*)alloc(32768, 2);
  bf16_t* Wd1T = (bf16_t*)alloc(131072, 2);
  bf16_t* Wd2T = (bf16_t*)alloc(32768, 2);
  float*  gridt= (float*) alloc(262144, 4);    // [2048,128]
  bf16_t* qc   = (bf16_t*)alloc(16777216, 2);  // [B*2048,256]
  bf16_t* hb   = (bf16_t*)alloc(4194304, 2);   // [B*2048,64]
  bf16_t* lb   = (bf16_t*)alloc(4194304, 2);   // [B*2048,64]
  bf16_t* gT   = (bf16_t*)alloc(16777216, 2);  // [B,256,2048]
  bf16_t* ao   = (bf16_t*)alloc(16777216, 2);  // [B*2048,256]
  bf16_t* p1b  = (bf16_t*)alloc(8388608, 2);   // p1 as bf16, viewed [B*512,512]
  bf16_t* d1   = (bf16_t*)alloc(4194304, 2);   // [B*512,256]
  float*  dn   = (float*) alloc(2097152, 4);   // down(p1)  [B*512,128]

  // precompute: bf16 conversions, weight transposes, grid term
  k_f32_to_bf16<<<cdivu(2097152,256),256,0,stream>>>(p, pb, 2097152);
  k_transpose<<<cdivu(128*128,256),256,0,stream>>>(Wup, WupT, 128, 128);
  k_transpose<<<cdivu(256*64,256),256,0,stream>>>(Wh, WhT, 256, 64);
  k_transpose<<<cdivu(256*64,256),256,0,stream>>>(Wl, WlT, 256, 64);
  k_transpose<<<cdivu(256*256,256),256,0,stream>>>(Wg, WgT, 256, 256);
  k_transpose<<<cdivu(256*128,256),256,0,stream>>>(Wf, WfT, 256, 128);
  k_transpose<<<cdivu(512*256,256),256,0,stream>>>(Wd1, Wd1T, 512, 256);
  k_transpose<<<cdivu(256*128,256),256,0,stream>>>(Wd2, Wd2T, 256, 128);
  k_gridterm<<<cdivu(NQ*CH,256),256,0,stream>>>(Wup, bup, gridt);

  auto run_up = [&](const bf16_t* pin, int accum, bf16_t* p1out){
    const long M = (long)NB*NQ;                 // 65536
    k_qc_copy<<<cdivu(M*CH,256),256,0,stream>>>(pin, qc);
    // q2d = relu(q @ Wup[:128] + gridterm)  -> qc[:,128:256]
    wmma_gemm<<<dim3(cdivu(M,256),2),256,0,stream>>>(qc,256, WupT,128,
        nullptr, gridt,128, qc+128,256, nullptr,0, nullptr,0,0, 1);
    // h, l, g
    wmma_gemm<<<dim3(cdivu(M,256),1),256,0,stream>>>(qc,256, WhT,256,
        bh, nullptr,0, hb,64, nullptr,0, nullptr,0,0, 1);
    wmma_gemm<<<dim3(cdivu(M,256),1),256,0,stream>>>(qc,256, WlT,256,
        bl, nullptr,0, lb,64, nullptr,0, nullptr,0,0, 1);
    wmma_gemm<<<dim3(cdivu(M,256),4),256,0,stream>>>(qc,256, WgT,256,
        bg, nullptr,0, nullptr,0, gT,256, nullptr,0,0, 1);
    // ao = qc + softmax(l h^T) g
    attn_kernel<<<NB*(NQ/128),256,0,stream>>>(lb, hb, gT, qc, ao);
    // p1 = relu(ao @ Wf + bf) -> f32 d_out (store/accum) + bf16 copy
    wmma_gemm<<<dim3(cdivu(M,256),2),256,0,stream>>>(ao,256, WfT,256,
        bfv, nullptr,0, p1out,128, nullptr,0, out,128,accum, 1);
  };

  run_up(pb, /*store*/0, p1b);                  // p1 -> d_out, p1b
  // down(p1): [B,512,512] @ Wd1 -> relu -> @ Wd2 -> relu
  wmma_gemm<<<dim3(cdivu((long)NB*NIN,256),4),256,0,stream>>>(p1b,512, Wd1T,512,
      bd1, nullptr,0, d1,256, nullptr,0, nullptr,0,0, 1);
  wmma_gemm<<<dim3(cdivu((long)NB*NIN,256),2),256,0,stream>>>(d1,256, Wd2T,256,
      bd2, nullptr,0, nullptr,0, nullptr,0, dn,128,0, 1);
  k_sub_bf16<<<cdivu(2097152,256),256,0,stream>>>(p, dn, pb2, 2097152);
  run_up(pb2, /*accumulate p2*/1, nullptr);     // d_out += p2
}
